// FactorizedFNO2d_45406394253957
// MI455X (gfx1250) — matrix operand order repaired and checked by reading, not compiled, not run on hardware
//
#include <hip/hip_runtime.h>
#include <cmath>

// ---------------------------------------------------------------------------
// FactorizedFNO2d forward for MI455X (gfx1250, wave32, WMMA).
// All heavy math is v_wmma_f32_16x16x32_f16 tiles; activations stored f16
// (memory-bound problem: halve HBM bytes, keep 143MB->71MB tensor L2-resident),
// accumulation in f32. Truncated rFFT/irFFT folded into GEMMs vs precomputed
// DFT matrices.
// ---------------------------------------------------------------------------

typedef __attribute__((ext_vector_type(16))) _Float16 v16h;
typedef __attribute__((ext_vector_type(8)))  _Float16 v8h;
typedef __attribute__((ext_vector_type(8)))  float    v8f;

#define HP   264       // padded spatial extent (256 + PAD 8)
#define LDY  288       // row stride (264 padded up to multiple of 32)
#define RES  256
#define CW_  64        // channel width

__device__ __forceinline__ v8f wmma_f16(v16h a, v16h b, v8f c) {
  // 8 args: (neg_a, A, neg_b, B, c_mod, C, reuse_a, reuse_b)
  return __builtin_amdgcn_wmma_f32_16x16x32_f16(false, a, false, b, (short)0, c,
                                                false, false);
}

// A fragment: 16x32 f16, row-major source, contiguous K.
// lane: row = lane%16, half = lane/16 holds K = 8h..8h+7 and 16+8h..16+8h+7.
__device__ __forceinline__ v16h frag_a(const _Float16* A, int lda, int k0, int lane) {
  int r = lane & 15, h = lane >> 4;
  const _Float16* p = A + (size_t)r * lda + k0 + (h << 3);
  v8h lo = *(const v8h*)p;
  v8h hi = *(const v8h*)(p + 16);
  v16h f;
#pragma unroll
  for (int e = 0; e < 8; ++e) { f[e] = lo[e]; f[e + 8] = hi[e]; }
  return f;
}

// B fragment: 32x16 f16 from transposed (N-major) storage Bt[n][k].
// lane: n = lane%16, half = lane/16 holds K = 8h..8h+7 and 16+8h..16+8h+7.
__device__ __forceinline__ v16h frag_b(const _Float16* Bt, int ldb, int k0, int lane) {
  int n = lane & 15, h = lane >> 4;
  const _Float16* p = Bt + (size_t)n * ldb + k0 + (h << 3);
  v8h lo = *(const v8h*)p;
  v8h hi = *(const v8h*)(p + 16);
  v16h f;
#pragma unroll
  for (int e = 0; e < 8; ++e) { f[e] = lo[e]; f[e + 8] = hi[e]; }
  return f;
}

// ---------------------------------------------------------------------------
// Weight preparation (f32 -> f16, layout permutes). Tiny, runs every call.
// ---------------------------------------------------------------------------
__global__ void k_prep_weights(const float* __restrict__ W2, const float* __restrict__ Wq1,
                               const float* __restrict__ cw,
                               const float* __restrict__ wxre, const float* __restrict__ wxim,
                               const float* __restrict__ wyre, const float* __restrict__ wyim,
                               _Float16* W2T, _Float16* Wq1T, _Float16* CWf,
                               _Float16* WxR, _Float16* WxI, _Float16* WxIN,
                               _Float16* WyR, _Float16* WyI, _Float16* WyIN) {
  int t = blockIdx.x * 256 + threadIdx.x;
  if (t < 8192) {               // W2 (128x64) -> W2T[o][k], ldb=128
    int o = t >> 7, k = t & 127;
    W2T[t] = (_Float16)W2[k * 64 + o];
  }
  if (t < 8192) {               // Wq1 (64x128) -> Wq1T[j][i], ldb=64
    int j = t >> 6, i = t & 63;
    Wq1T[t] = (_Float16)Wq1[i * 128 + j];
  }
  if (t < 16384) CWf[t] = (_Float16)cw[t];   // conv_w (4,64,64) [l][o][i] as-is
  if (t < 262144) {             // spectral weights -> [l][m][o][i] (+ negated Im)
    int l = t >> 16;
    int r = t & 65535;
    int m = r >> 12;
    int o = (r >> 6) & 63;
    int i = r & 63;
    int src = ((l * 64 + i) * 64 + o) * 16 + m;
    float xr = wxre[src], xi = wxim[src], yr = wyre[src], yi = wyim[src];
    WxR[t] = (_Float16)xr; WxI[t] = (_Float16)xi; WxIN[t] = (_Float16)(-xi);
    WyR[t] = (_Float16)yr; WyI[t] = (_Float16)yi; WyIN[t] = (_Float16)(-yi);
  }
}

// Truncated DFT matrices. FtF[j][n] (32x288): j<16 -> cos, j>=16 -> -sin
// (forward rFFT, 16 modes). GtF[y][j] (272x32): irfft with 1/N and 2x for m>0.
__global__ void k_prep_dft(_Float16* FtF, _Float16* GtF) {
  int t = blockIdx.x * 256 + threadIdx.x;
  const float W0 = 6.28318530717958647692f / 264.0f;
  if (t < 32 * 288) {
    int j = t / 288, n = t % 288;
    float v = 0.f;
    if (n < HP) {
      int m = j & 15;
      int mn = (m * n) % 264;
      float th = W0 * (float)mn;
      v = (j < 16) ? cosf(th) : -sinf(th);
    }
    FtF[t] = (_Float16)v;
  }
  if (t < 272 * 32) {
    int y = t >> 5, j = t & 31;
    float v = 0.f;
    if (y < HP) {
      int m = j & 15;
      float c = ((m == 0) ? 1.f : 2.f) / 264.f;
      int mn = (m * y) % 264;
      float th = W0 * (float)mn;
      v = (j < 16) ? c * cosf(th) : -c * sinf(th);
    }
    GtF[t] = (_Float16)v;
  }
}

// ---------------------------------------------------------------------------
// Fused lifting: per 16-pixel tile, scalar 3->128 MLP into LDS, then WMMA
// 128->64 against W2T, writing channels-last activation CL[b][x][y][c].
// ---------------------------------------------------------------------------
__global__ void k_lift(const float* __restrict__ x_in, const float* __restrict__ W1,
                       const float* __restrict__ b1, const _Float16* __restrict__ W2T,
                       const float* __restrict__ b2, _Float16* __restrict__ CL) {
  __shared__ _Float16 Hs[16 * 128];
  int tid = threadIdx.x;
  int p0 = blockIdx.x << 4;               // 256-grid pixel base (b,x,y packed 3/8/8)
  {
    int row = tid >> 4;
    int j0 = (tid & 15) << 3;
    int p = p0 + row;
    int b = p >> 16, x = (p >> 8) & 255, y = p & 255;
    float gy = (float)x * (1.0f / 255.0f);
    float gx = (float)y * (1.0f / 255.0f);
    float xv = x_in[(size_t)b * 65536 + x * 256 + y];
#pragma unroll
    for (int u = 0; u < 8; ++u) {
      int j = j0 + u;
      float v = gy * W1[j] + gx * W1[128 + j] + xv * W1[256 + j] + b1[j];
      v = v > 0.f ? v : 0.01f * v;        // leaky relu
      Hs[row * 128 + j] = (_Float16)v;
    }
  }
  __syncthreads();
  int wave = tid >> 5, lane = tid & 31;
  if (wave < 4) {                          // N=64 -> 4 column tiles
    const _Float16* A = (const _Float16*)Hs;
    const _Float16* B = W2T + wave * 16 * 128;
    v8f acc = {};
#pragma unroll
    for (int k0 = 0; k0 < 128; k0 += 32) {
      v16h a = frag_a(A, 128, k0, lane);
      v16h b = frag_b(B, 128, k0, lane);
      acc = wmma_f16(a, b, acc);
    }
    int n = lane & 15, h = lane >> 4;
    int o = wave * 16 + n;
    float bias = b2[o];
#pragma unroll
    for (int v = 0; v < 8; ++v) {
      int p = p0 + v + (h << 3);
      int b = p >> 16, x = (p >> 8) & 255, y = p & 255;
      CL[((size_t)(b * HP + x) * HP + y) * 64 + o] = (_Float16)(acc[v] + bias);
    }
  }
}

// ---------------------------------------------------------------------------
// Pointwise conv: ACC[b,o,x,y] = sum_i W[o,i] * act[b,i,x,y]. A = conv_w
// (M=64 rows o), Bt = channels-last activation (contiguous K=i). First writer
// of ACC (plain store, y-coalesced across lanes).
// ---------------------------------------------------------------------------
__global__ void k_conv(const _Float16* __restrict__ Wc, const _Float16* __restrict__ CL,
                       float* __restrict__ ACC) {
  int lane = threadIdx.x & 31, wave = threadIdx.x >> 5;
  int tile = blockIdx.x * 8 + wave;        // 4 Mtiles * 34848 Ntiles = 139392
  int nt = tile >> 2, mt = tile & 3;
  const _Float16* A = Wc + mt * 16 * 64;
  const _Float16* B = CL + (size_t)nt * 16 * 64;
  v8f acc = {};
#pragma unroll
  for (int k0 = 0; k0 < 64; k0 += 32) {
    v16h a = frag_a(A, 64, k0, lane);
    v16h b = frag_b(B, 64, k0, lane);
    acc = wmma_f16(a, b, acc);
  }
  int n = lane & 15, h = lane >> 4;
  int p = nt * 16 + n;                     // pixel (b,x,y) over 264-grid
  int b = p / 69696;
  int q = p % 69696;
  int x = q / HP, y = q % HP;
#pragma unroll
  for (int v = 0; v < 8; ++v) {
    int o = mt * 16 + v + (h << 3);
    ACC[((size_t)(b * 64 + o) * HP + x) * LDY + y] = acc[v];
  }
}

// ---------------------------------------------------------------------------
// Stage A: forward truncated DFT. C[(b,c,s)][j] = sum_k act[row][k]*FtF[j][k].
// Scatter to mode-major Q[m][b][s][c] so stage B has contiguous K=c.
// ---------------------------------------------------------------------------
__global__ void k_stageA(const _Float16* __restrict__ act, const _Float16* __restrict__ FtF,
                         _Float16* __restrict__ QRe, _Float16* __restrict__ QIm) {
  int lane = threadIdx.x & 31, wave = threadIdx.x >> 5;
  int tile = blockIdx.x * 8 + wave;        // 8448 row-tiles * 2 col-tiles
  int rt = tile >> 1, ct = tile & 1;
  const _Float16* A = act + (size_t)rt * 16 * LDY;
  const _Float16* B = FtF + ct * 16 * LDY;
  const _Float16* pf = A + (size_t)(lane & 15) * LDY + ((lane >> 4) << 3);
  v8f acc = {};
#pragma unroll
  for (int k0 = 0; k0 < LDY; k0 += 32) {
    if (k0 + 32 < LDY) __builtin_prefetch(pf + k0 + 32, 0, 1);  // global_prefetch
    v16h a = frag_a(A, LDY, k0, lane);
    v16h b = frag_b(B, LDY, k0, lane);
    acc = wmma_f16(a, b, acc);
  }
  int n = lane & 15, h = lane >> 4;
  int j = ct * 16 + n;
  int m = j & 15;
  _Float16* Qb = (j < 16 ? QRe : QIm) + (size_t)m * (8 * HP * 64);
#pragma unroll
  for (int v = 0; v < 8; ++v) {
    int r = rt * 16 + v + (h << 3);        // flat (b,c,s)
    int b = r / 16896;                     // 64*264
    int rem = r % 16896;
    int c = rem / HP, s = rem % HP;
    Qb[(size_t)(b * HP + s) * 64 + c] = (_Float16)acc[v];
  }
}

// ---------------------------------------------------------------------------
// Stage B: per-mode complex 64x64 channel mix.
//   Rr = QRe*Wr + QIm*(-Wi) ;  Ri = QRe*Wi + QIm*Wr
// (f16 WMMA has no A-negate per ISA NEG table, so -Wi is pre-materialized.)
// Output S[b][s][o][32] (Re at j=m, Im at j=16+m) for stage C contiguous-K.
// ---------------------------------------------------------------------------
__global__ void k_stageB(const _Float16* __restrict__ QRe, const _Float16* __restrict__ QIm,
                         const _Float16* __restrict__ Wr, const _Float16* __restrict__ Wi,
                         const _Float16* __restrict__ WiN, _Float16* __restrict__ S) {
  int lane = threadIdx.x & 31, wave = threadIdx.x >> 5;
  int tile = blockIdx.x * 8 + wave;        // 16 m * 132 rt * 4 ct = 8448
  int m = tile / 528;
  int rem = tile % 528;
  int rt = rem >> 2, ct = rem & 3;
  const _Float16* Ar = QRe + (size_t)m * (8 * HP * 64) + (size_t)rt * 16 * 64;
  const _Float16* Ai = QIm + (size_t)m * (8 * HP * 64) + (size_t)rt * 16 * 64;
  const _Float16* Br = Wr + (size_t)m * 4096 + ct * 16 * 64;
  const _Float16* Bi = Wi + (size_t)m * 4096 + ct * 16 * 64;
  const _Float16* Bn = WiN + (size_t)m * 4096 + ct * 16 * 64;
  v8f ar = {}, ai = {};
#pragma unroll
  for (int k0 = 0; k0 < 64; k0 += 32) {
    v16h fr = frag_a(Ar, 64, k0, lane);
    v16h fi = frag_a(Ai, 64, k0, lane);
    v16h br = frag_b(Br, 64, k0, lane);
    v16h bi = frag_b(Bi, 64, k0, lane);
    v16h bn = frag_b(Bn, 64, k0, lane);
    ar = wmma_f16(fr, br, ar);
    ar = wmma_f16(fi, bn, ar);
    ai = wmma_f16(fr, bi, ai);
    ai = wmma_f16(fi, br, ai);
  }
  int n = lane & 15, h = lane >> 4;
  int o = ct * 16 + n;
#pragma unroll
  for (int v = 0; v < 8; ++v) {
    int rf = rt * 16 + v + (h << 3);       // (b,s) flat
    _Float16* p = S + ((size_t)rf * 64 + o) * 32;
    p[m] = (_Float16)ar[v];
    p[16 + m] = (_Float16)ai[v];
  }
}

// ---------------------------------------------------------------------------
// Stage C: inverse truncated DFT; accumulate into f32 ACC.
// dir=0 (y-transform): row=(b,s=x,o), col=y. dir=1 (x, transposed space):
// row=(b,s=y,o), col=x (guard x<264 to avoid crossing planes).
// ---------------------------------------------------------------------------
__global__ void k_stageC(const _Float16* __restrict__ S, const _Float16* __restrict__ GtF,
                         float* __restrict__ ACC, int dir) {
  int lane = threadIdx.x & 31, wave = threadIdx.x >> 5;
  int tile = blockIdx.x * 8 + wave;        // 8448 rt * 17 ct = 143616
  int rt = tile / 17, ct = tile % 17;
  const _Float16* A = S + (size_t)rt * 16 * 32;
  const _Float16* B = GtF + ct * 16 * 32;
  v8f acc = {};
  {
    v16h a = frag_a(A, 32, 0, lane);
    v16h b = frag_b(B, 32, 0, lane);
    acc = wmma_f16(a, b, acc);
  }
  int n = lane & 15, h = lane >> 4;
  int col = ct * 16 + n;
#pragma unroll
  for (int v = 0; v < 8; ++v) {
    int r = rt * 16 + v + (h << 3);
    int b = r / 16896;
    int q = r % 16896;
    int s = q >> 6, o = q & 63;
    if (dir == 0) {
      ACC[((size_t)(b * 64 + o) * HP + s) * LDY + col] += acc[v];  // col<272<LDY
    } else {
      if (col < HP)
        ACC[((size_t)(b * 64 + o) * HP + col) * LDY + s] += acc[v];
    }
  }
}

// ACC + conv bias (+ leaky relu except last layer) -> f16 activation.
__global__ void k_finalize(const float* __restrict__ ACC, const float* __restrict__ cb,
                           _Float16* __restrict__ act, int l, int relu) {
  int t = blockIdx.x * 256 + threadIdx.x;
  if (t >= 8 * 64 * HP * HP) return;
  int y = t % HP;
  int row = t / HP;                        // (b*64+o)*264 + x
  int o = (row / HP) & 63;
  float v = ACC[(size_t)row * LDY + y] + cb[l * 64 + o];
  if (relu) v = v > 0.f ? v : 0.01f * v;
  act[(size_t)row * LDY + y] = (_Float16)v;
}

// Generic LDS-tiled f16 slab transpose: dst[col][row] = src[row][col] per slab.
// slab s: b=s/D, r2=s%D; src base = b*sA + r2*sB; dst base = b*dA + r2*dB.
__global__ void k_transpose(const _Float16* __restrict__ src, _Float16* __restrict__ dst,
                            int rows, int cols, int srs, int drs, int D,
                            size_t sA, size_t sB, size_t dA, size_t dB) {
  __shared__ _Float16 tile[16][17];
  int s = blockIdx.z;
  int b = s / D, r2 = s % D;
  const _Float16* Sp = src + (size_t)b * sA + (size_t)r2 * sB;
  _Float16* Dp = dst + (size_t)b * dA + (size_t)r2 * dB;
  int tx = threadIdx.x & 15, ty = threadIdx.x >> 4;
  int c0 = blockIdx.x << 4, r0 = blockIdx.y << 4;
  int r = r0 + ty, c = c0 + tx;
  tile[ty][tx] = (r < rows && c < cols) ? Sp[(size_t)r * srs + c] : (_Float16)0.f;
  __syncthreads();
  int dc = c0 + ty, dr = r0 + tx;
  if (dc < cols && dr < rows) Dp[(size_t)dc * drs + dr] = tile[tx][ty];
}

// ---------------------------------------------------------------------------
// Fused projection: WMMA 64->128 (with bq1) into LDS, leaky relu + 128->1 dot
// (Wq2) scalar tail, write fp32 output on the 256x256 crop.
// ---------------------------------------------------------------------------
__global__ void k_proj(const _Float16* __restrict__ CL, const _Float16* __restrict__ Wq1T,
                       const float* __restrict__ bq1, const float* __restrict__ Wq2,
                       const float* __restrict__ bq2, float* __restrict__ out) {
  __shared__ float Hq[16 * 128];
  int tid = threadIdx.x, wave = tid >> 5, lane = tid & 31;
  int p0 = blockIdx.x << 4;                // 256-grid pixel base; y-tile never splits
  int b = p0 >> 16, x = (p0 >> 8) & 255, y0 = p0 & 255;
  size_t rowbase = (size_t)(b * HP + x) * HP + y0;
  const _Float16* A = CL + rowbase * 64;
  const _Float16* B = Wq1T + wave * 16 * 64;
  v8f acc = {};
#pragma unroll
  for (int k0 = 0; k0 < 64; k0 += 32) {
    v16h a = frag_a(A, 64, k0, lane);
    v16h bb = frag_b(B, 64, k0, lane);
    acc = wmma_f16(a, bb, acc);
  }
  int n = lane & 15, h = lane >> 4;
  int j = wave * 16 + n;
  float bias = bq1[j];
#pragma unroll
  for (int v = 0; v < 8; ++v) Hq[(v + (h << 3)) * 128 + j] = acc[v] + bias;
  __syncthreads();
  if (tid < 16) {
    float sacc = bq2[0];
#pragma unroll 8
    for (int jj = 0; jj < 128; ++jj) {
      float hv = Hq[tid * 128 + jj];
      hv = hv > 0.f ? hv : 0.01f * hv;
      sacc += hv * Wq2[jj];
    }
    out[(size_t)b * 65536 + x * 256 + y0 + tid] = sacc;
  }
}

// ---------------------------------------------------------------------------
extern "C" void kernel_launch(void* const* d_in, const int* in_sizes, int n_in,
                              void* d_out, int out_size, void* d_ws, size_t ws_size,
                              hipStream_t stream) {
  (void)in_sizes; (void)n_in; (void)out_size; (void)ws_size;
  const float* x_in = (const float*)d_in[0];
  const float* W1   = (const float*)d_in[1];
  const float* b1   = (const float*)d_in[2];
  const float* W2   = (const float*)d_in[3];
  const float* b2   = (const float*)d_in[4];
  const float* cw   = (const float*)d_in[5];
  const float* cb   = (const float*)d_in[6];
  const float* wxre = (const float*)d_in[7];
  const float* wxim = (const float*)d_in[8];
  const float* wyre = (const float*)d_in[9];
  const float* wyim = (const float*)d_in[10];
  const float* Wq1  = (const float*)d_in[11];
  const float* bq1  = (const float*)d_in[12];
  const float* Wq2  = (const float*)d_in[13];
  const float* bq2  = (const float*)d_in[14];

  char* ws = (char*)d_ws;
  size_t off = 0;
  auto carve = [&](size_t bytes) -> char* {
    char* p = ws + off;
    off += (bytes + 255) & ~(size_t)255;
    return p;
  };
  _Float16* ACT  = (_Float16*)carve(77856768);   // [b][c][x][288] f16
  _Float16* ACTT = (_Float16*)carve(77856768);   // [b][c][y][288] f16
  _Float16* CL   = (_Float16*)carve(71368704);   // [b][x][y][c]   f16
  float*    ACC  = (float*)   carve(155713536);  // [b][c][x][288] f32 accumulator
  _Float16* QRe  = (_Float16*)carve(4325376);    // [m][b][s][c]
  _Float16* QIm  = (_Float16*)carve(4325376);
  _Float16* Smix = (_Float16*)carve(8650752);    // [b][s][o][32]
  _Float16* FtF  = (_Float16*)carve(18432);      // [32][288]
  _Float16* GtF  = (_Float16*)carve(17408);      // [272][32]
  _Float16* W2T  = (_Float16*)carve(16384);
  _Float16* Wq1T = (_Float16*)carve(16384);
  _Float16* CWf  = (_Float16*)carve(32768);
  _Float16* WxR  = (_Float16*)carve(524288);
  _Float16* WxI  = (_Float16*)carve(524288);
  _Float16* WxIN = (_Float16*)carve(524288);
  _Float16* WyR  = (_Float16*)carve(524288);
  _Float16* WyI  = (_Float16*)carve(524288);
  _Float16* WyIN = (_Float16*)carve(524288);

  hipMemsetAsync(CL, 0, 71368704, stream);       // zero-pad region for lift
  k_prep_weights<<<1024, 256, 0, stream>>>(W2, Wq1, cw, wxre, wxim, wyre, wyim,
                                           W2T, Wq1T, CWf, WxR, WxI, WxIN,
                                           WyR, WyI, WyIN);
  k_prep_dft<<<36, 256, 0, stream>>>(FtF, GtF);
  k_lift<<<32768, 256, 0, stream>>>(x_in, W1, b1, W2T, b2, CL);
  // CL -> ACT (per (b,x) slab: 264y x 64c -> 64c x 264y)
  k_transpose<<<dim3(4, 17, 2112), 256, 0, stream>>>(
      CL, ACT, 264, 64, 64, 76032, 264,
      (size_t)4460544, (size_t)16896, (size_t)4866048, (size_t)288);
  // ACT -> ACTT (per (b,c) slab: 264x x 264y -> 264y x 264x)
  k_transpose<<<dim3(17, 17, 512), 256, 0, stream>>>(
      ACT, ACTT, 264, 264, 288, 288, 512,
      (size_t)0, (size_t)76032, (size_t)0, (size_t)76032);

  for (int l = 0; l < 4; ++l) {
    k_conv<<<17424, 256, 0, stream>>>(CWf + l * 4096, CL, ACC);
    // y-direction spectral path
    k_stageA<<<2112, 256, 0, stream>>>(ACT, FtF, QRe, QIm);
    k_stageB<<<1056, 256, 0, stream>>>(QRe, QIm, WyR + l * 65536, WyI + l * 65536,
                                       WyIN + l * 65536, Smix);
    k_stageC<<<17952, 256, 0, stream>>>(Smix, GtF, ACC, 0);
    // x-direction spectral path (on transposed activation)
    k_stageA<<<2112, 256, 0, stream>>>(ACTT, FtF, QRe, QIm);
    k_stageB<<<1056, 256, 0, stream>>>(QRe, QIm, WxR + l * 65536, WxI + l * 65536,
                                       WxIN + l * 65536, Smix);
    k_stageC<<<17952, 256, 0, stream>>>(Smix, GtF, ACC, 1);
    k_finalize<<<139392, 256, 0, stream>>>(ACC, cb, ACT, l, (l < 3) ? 1 : 0);
    // rebuild channels-last view (conv of next layer / projection)
    k_transpose<<<dim3(17, 4, 2112), 256, 0, stream>>>(
        ACT, CL, 64, 264, 76032, 64, 264,
        (size_t)4866048, (size_t)288, (size_t)4460544, (size_t)16896);
    if (l < 3)
      k_transpose<<<dim3(17, 17, 512), 256, 0, stream>>>(
          ACT, ACTT, 264, 264, 288, 288, 512,
          (size_t)0, (size_t)76032, (size_t)0, (size_t)76032);
  }
  k_proj<<<32768, 256, 0, stream>>>(CL, Wq1T, bq1, Wq2, bq2, (float*)d_out);
}
